// Model_32719060861079
// MI455X (gfx1250) — matrix operand order repaired and checked by reading, not compiled, not run on hardware
//
#include <hip/hip_runtime.h>
#include <math.h>

// MI455X (gfx1250): factorized Cooley-Tukey DFTs as complex GEMMs on
// v_wmma_f32_16x16x32_f16. LDS tiles live in WMMA fragment order (2x
// ds_load_b128 per operand); zero-padded forward transforms shorten the
// K-reduction (Keff=128 of 256) instead of masking loads — padding chunks are
// uniform, so no divergence and half the GEMM work. Twiddles come from an
// exact mod-N f32 table (sign always -1). Wave32, 8 waves/block.

typedef __attribute__((ext_vector_type(16))) _Float16 v16h;
typedef __attribute__((ext_vector_type(8)))  float    v8f;

#define B_       4
#define NEV      32
#define NH       16
#define NSAMP    32768
#define NFRAMES  128
#define NPAD     65536
#define NROOMS   8
#define MIN_F    (20.0f/11025.0f)
#define MAX_F    (4000.0f/11025.0f)
#define LSQ2PI   0.91893853320467274178f
#define TWO_PI   6.28318530717958647692f

struct EvParams { float fmu,fsig,fnorm,tmu,tsig,tnorm,amp; float freqs[NH]; };

union PackU { _Float16 h[2]; unsigned u; };

// ------- tables: f16 DFT matrices (symmetric) + f32 twiddle tables ----------
__global__ void k_tables(_Float16* t256re, _Float16* t256im,
                         _Float16* t128re, _Float16* t128im,
                         float* tw32, float* tw64) {
  int t = blockIdx.x*256 + threadIdx.x;
  if (t < 256*256) {
    int n = t >> 8, k = t & 255; int m = (n*k) & 255;
    float th = -TWO_PI * (float)m * (1.0f/256.0f);
    t256re[t] = (_Float16)cosf(th); t256im[t] = (_Float16)sinf(th);
  }
  if (t < 128*128) {
    int n = t >> 7, k = t & 127; int m = (n*k) & 127;
    float th = -TWO_PI * (float)m * (1.0f/128.0f);
    t128re[t] = (_Float16)cosf(th); t128im[t] = (_Float16)sinf(th);
  }
  if (t < NPAD) {
    float th = -TWO_PI * (float)t * (1.0f/(float)NPAD);
    tw64[2*t] = cosf(th); tw64[2*t+1] = sinf(th);
  }
  if (t < NSAMP) {
    float th = -TWO_PI * (float)t * (1.0f/(float)NSAMP);
    tw32[2*t] = cosf(th); tw32[2*t+1] = sinf(th);
  }
}

// ------- per-event parameter derivation (all analytic) ----------------------
__global__ void k_params(const float* __restrict__ ep, const float* __restrict__ mx_raw,
                         const float* __restrict__ room_logits,
                         EvParams* __restrict__ P, float* __restrict__ resfr,
                         float* __restrict__ mx, float* __restrict__ rooms) {
  int t = blockIdx.x*blockDim.x + threadIdx.x;
  if (t < B_*NEV*NH) {
    int h = t & 15; int be = t >> 4;
    float r = ep[be*22 + 6 + h]*0.2f + 0.8f;
    float c = 1.0f;
    for (int fr = 0; fr < NFRAMES; ++fr) { c *= r; resfr[(size_t)t*NFRAMES + fr] = c; }
  }
  if (t < B_*NEV) {
    const float* e = ep + t*22;
    EvParams p;
    float fm = e[0]*e[0];
    p.fmu = MIN_F + fm*(MAX_F - MIN_F); p.fsig = 1e-8f + e[1]*0.1f;
    int fi = (int)roundf(p.fmu*16384.0f); fi = fi < 0 ? 0 : (fi > 16384 ? 16384 : fi);
    float rr = (float)fi*(1.0f/16384.0f);
    float d  = (rr - p.fmu)/p.fsig;
    p.fnorm = 1.0f/(expf(-0.5f*d*d - logf(p.fsig) - LSQ2PI) + 1e-8f);
    p.tmu = e[2]; p.tsig = 1e-8f + e[3]*0.1f;
    int ti = (int)roundf(p.tmu*32767.0f); ti = ti < 0 ? 0 : (ti > 32767 ? 32767 : ti);
    float tr = (float)ti*(1.0f/32767.0f);
    float dt = (tr - p.tmu)/p.tsig;
    p.tnorm = 1.0f/(expf(-0.5f*dt*dt - logf(p.tsig) - LSQ2PI) + 1e-8f);
    float f0 = e[4]*e[4]; p.amp = e[5]*e[5];
    float f0s = MIN_F + (MAX_F - MIN_F)*f0;
    for (int h = 0; h < NH; ++h) { float f = f0s*(float)(h+1); p.freqs[h] = (f >= 1.0f) ? 0.0f : f; }
    P[t] = p;
  }
  if (t < B_) {
    mx[t] = 1.0f/(1.0f + expf(-mx_raw[t]));
    float m = -1e30f;
    for (int r = 0; r < NROOMS; ++r) m = fmaxf(m, room_logits[t*NROOMS + r]);
    float s = 0.0f, ev[NROOMS];
    for (int r = 0; r < NROOMS; ++r) { ev[r] = expf(room_logits[t*NROOMS + r] - m); s += ev[r]; }
    for (int r = 0; r < NROOMS; ++r) rooms[t*NROOMS + r] = ev[r]/s;
  }
}

// ------- damped-harmonic resonance signals ----------------------------------
__global__ void k_resonance(const EvParams* __restrict__ P, const float* __restrict__ resfr,
                            float* __restrict__ res) {
  int be = blockIdx.x;
  int n  = blockIdx.y*256 + threadIdx.x;
  EvParams p = P[be];
  float fc = (n + 0.5f)*(1.0f/256.0f) - 0.5f;
  fc = fminf(fmaxf(fc, 0.0f), 127.0f);
  int lo = (int)fc; int hi = (lo + 1 > 127) ? 127 : lo + 1;
  float w = fc - (float)lo;
  const float* rf = resfr + (size_t)be*NH*NFRAMES;
  float s = 0.0f;
  for (int h = 0; h < NH; ++h) {
    float rv = rf[h*NFRAMES + lo]*(1.0f - w) + rf[h*NFRAMES + hi]*w;
    double ph = (double)p.freqs[h]*(double)(n + 1);   // exact phase, no cumsum drift
    ph -= floor(ph);
    s += rv * sinf((float)(TWO_PI*ph));
  }
  res[(size_t)be*NSAMP + n] = s;
}

// ------- misc pointwise kernels ---------------------------------------------
__global__ void k_rooms_mix(const float* __restrict__ rooms, const float* __restrict__ ir,
                            float* __restrict__ imp) {
  int idx = blockIdx.x*256 + threadIdx.x;
  int b = idx >> 15, n = idx & (NSAMP - 1);
  float s = 0.0f;
  for (int r = 0; r < NROOMS; ++r) s += rooms[b*NROOMS + r]*ir[(size_t)r*NSAMP + n];
  imp[(size_t)b*NSAMP + n] = s;
}

__global__ void k_specwin(const float* __restrict__ nspec, const EvParams* __restrict__ P,
                          float* __restrict__ spec32) {
  int e = blockIdx.x;
  int k = blockIdx.y*256 + threadIdx.x;
  EvParams p = P[e];
  int wi = (k <= 16384) ? k : (NSAMP - k);      // Hermitian mirror of the window
  float r = (float)wi*(1.0f/16384.0f);
  float d = (r - p.fmu)/p.fsig;
  float w = expf(-0.5f*d*d - logf(p.fsig) - LSQ2PI)*p.fnorm;
  size_t o = ((size_t)e*NSAMP + k)*2;
  spec32[o]     = nspec[2*k]*w;
  spec32[o + 1] = nspec[2*k + 1]*w;
}

__global__ void k_envelope(const float* __restrict__ bl, const EvParams* __restrict__ P,
                           float* __restrict__ loc) {
  int e = blockIdx.x;
  int n = blockIdx.y*256 + threadIdx.x;
  EvParams p = P[e];
  float r = (float)n*(1.0f/32767.0f);
  float d = (r - p.tmu)/p.tsig;
  float w = expf(-0.5f*d*d - logf(p.tsig) - LSQ2PI)*p.tnorm;
  size_t i = (size_t)e*NSAMP + n;
  loc[i] = bl[i]*w*p.amp;
}

__global__ void k_locsum(const float* __restrict__ loc, float* __restrict__ out) {
  int n = blockIdx.x*256 + threadIdx.x;
  float s = 0.0f;
  for (int e = 0; e < NEV; ++e) s += loc[(size_t)e*NSAMP + n];
  out[n] = s;
}

__global__ void k_accum(const float* __restrict__ Lh, const float* __restrict__ Rh,
                        float* __restrict__ acc) {
  int k = blockIdx.x*256 + threadIdx.x;
  float ar = 0.0f, ai = 0.0f;
  for (int e = 0; e < NEV; ++e) {
    size_t i = ((size_t)e*NPAD + k)*2;
    float lr = Lh[i], li = Lh[i+1], rr = Rh[i], ri = Rh[i+1];
    ar += lr*rr - li*ri; ai += lr*ri + li*rr;
  }
  acc[2*k] = ar; acc[2*k + 1] = ai;
}

__global__ void k_dry(const float* __restrict__ convs, const float* __restrict__ ls,
                      float* __restrict__ dry) {
  int n = blockIdx.x*256 + threadIdx.x;
  dry[n] = convs[n] + ls[n];
}

__global__ void k_wetspec(const float* __restrict__ Dh, const float* __restrict__ impspec,
                          float* __restrict__ Wh) {
  int k = blockIdx.x*256 + threadIdx.x;
  float dr = Dh[2*k], di = Dh[2*k+1], ir = impspec[2*k], ii = impspec[2*k+1];
  Wh[2*k] = dr*ir - di*ii; Wh[2*k+1] = dr*ii + di*ir;
}

__global__ void k_final(const float* __restrict__ dry, const float* __restrict__ wet,
                        const float* __restrict__ mxp, float* __restrict__ out) {
  int n = blockIdx.x*256 + threadIdx.x;
  float m = mxp[0];
  out[n] = dry[n]*(1.0f - m) + wet[n]*m;
}

// ------- WMMA complex GEMM DFT pass -----------------------------------------
// O[r][c] = Sum_{j<Keff} In[r][j] * W_L[j][c];  out stored at  out[r + M*c].
// IN_REAL inputs (rs==1, cs==M) are implicitly zero beyond kvalid columns;
// kvalid is a multiple of 32 so padded chunks are skipped wholesale (Keff).
template<bool IN_REAL, bool CONJ_IN, bool TWIDDLE, bool OUT_REAL>
__global__ void k_dft_pass(const float* __restrict__ in, float* __restrict__ out,
                           long in_stride, long out_stride, int M, int L,
                           const _Float16* __restrict__ dRe, const _Float16* __restrict__ dIm,
                           int kvalid, int in_rs, int in_cs,
                           const float* __restrict__ twTab, int tw_mask, float out_scale) {
  __shared__ _Float16 aRe[2][32][16] __attribute__((aligned(32)));
  __shared__ _Float16 aIm[2][32][16] __attribute__((aligned(32)));
  __shared__ _Float16 bRe[4][32][16] __attribute__((aligned(32)));
  __shared__ _Float16 bIm[4][32][16] __attribute__((aligned(32)));

  int sig  = blockIdx.z;
  int lane = threadIdx.x & 31;
  int wv   = threadIdx.x >> 5;
  int rt   = wv >> 2;          // 0..1  row tile
  int ct   = wv & 3;           // 0..3  col tile
  int rowB = blockIdx.x*32;
  int colB = blockIdx.y*64;
  const float* xin = in + (size_t)sig*in_stride;

  int Keff = IN_REAL ? kvalid : L;   // reduction shortened over implicit zeros

  v8f accP = {}, accQ = {}, accS = {};

  #pragma unroll 2
  for (int k0 = 0; k0 < Keff; k0 += 32) {
    // ---- stage A: 32 rows x 16 even-K pairs, packed f16 pair per store ----
    #pragma unroll
    for (int i = 0; i < 2; ++i) {
      int idx = threadIdx.x + 256*i;                 // 0..511
      int rr, q;
      if (TWIDDLE) { rr = idx & 31; q = idx >> 5; }  // pass-1: row-fast (rs==1)
      else         { q = idx & 15; rr = idx >> 4; }  // pass-2: K-fast (cs==1)
      int kk0 = 2*q;
      int ci0 = (rowB + rr)*in_rs + (k0 + kk0)*in_cs;  // fits in 32 bits
      float re0, im0, re1, im1;
      if (IN_REAL) {                                  // no bounds: chunk valid
        re0 = xin[ci0]; re1 = xin[ci0 + in_cs]; im0 = 0.0f; im1 = 0.0f;
      } else if (!TWIDDLE) {                          // pass-2: cs==1, b128 pair
        float4 cp = *(const float4*)(xin + 2*ci0);
        re0 = cp.x; im0 = cp.y; re1 = cp.z; im1 = cp.w;
      } else {                                        // inverse pass-1: cs==M
        float2 c0 = *(const float2*)(xin + 2*ci0);
        float2 c1 = *(const float2*)(xin + 2*(ci0 + in_cs));
        re0 = c0.x; im0 = CONJ_IN ? -c0.y : c0.y;
        re1 = c1.x; im1 = CONJ_IN ? -c1.y : c1.y;
      }
      // inverse of documented 16-bit A layout: (row,K) -> (lane, j)
      int art = rr >> 4, aMrow = rr & 15;
      int km  = kk0 & 15; int hi = (km >= 8);
      int basev = hi ? (km - 8) : km;
      int v = (kk0 < 16) ? (basev >> 1) : (4 + (basev >> 1));
      int alane = aMrow + (hi ? 16 : 0);
      PackU pr, pi;
      pr.h[0] = (_Float16)re0; pr.h[1] = (_Float16)re1;
      pi.h[0] = (_Float16)im0; pi.h[1] = (_Float16)im1;
      *(unsigned*)&aRe[art][alane][2*v] = pr.u;
      *(unsigned*)&aIm[art][alane][2*v] = pi.u;
    }
    // ---- stage B: symmetric DFT table read K-contiguous -> packed pair ----
    #pragma unroll
    for (int i = 0; i < 4; ++i) {
      int idx = threadIdx.x + 256*i;                 // 0..1023
      int q  = idx & 15;                             // K pair, fast -> coalesced
      int cc = idx >> 4;                             // 0..63
      int kk0 = 2*q;
      int ti = (colB + cc)*L + (k0 + kk0);           // W symmetric: [c][k]
      unsigned pu_re = *(const unsigned*)(dRe + ti);
      unsigned pu_im = *(const unsigned*)(dIm + ti);
      int bct = cc >> 4, bN = cc & 15;
      int blane = bN + ((kk0 & 8) ? 16 : 0);
      int j = (kk0 & 7) + ((kk0 >= 16) ? 8 : 0);
      *(unsigned*)&bRe[bct][blane][j] = pu_re;
      *(unsigned*)&bIm[bct][blane][j] = pu_im;
    }
    if (k0 + 32 < Keff) {  // prefetch next twiddle chunk (global_prefetch_b8)
      __builtin_prefetch((const void*)(dRe + colB*L + k0 + 32), 0, 1);
      __builtin_prefetch((const void*)(dIm + colB*L + k0 + 32), 0, 1);
    }
    __syncthreads();

    v16h are = *(const v16h*)&aRe[rt][lane][0];
    v16h aim = *(const v16h*)&aIm[rt][lane][0];
    v16h bre = *(const v16h*)&bRe[ct][lane][0];
    v16h bim = *(const v16h*)&bIm[ct][lane][0];

    accP = __builtin_amdgcn_wmma_f32_16x16x32_f16(false, are, false, bre, (short)0, accP, false, false);
    accQ = __builtin_amdgcn_wmma_f32_16x16x32_f16(false, aim, false, bim, (short)0, accQ, false, false);
    accS = __builtin_amdgcn_wmma_f32_16x16x32_f16(false, are, false, bim, (short)0, accS, false, false);
    accS = __builtin_amdgcn_wmma_f32_16x16x32_f16(false, aim, false, bre, (short)0, accS, false, false);
    __syncthreads();
  }

  // ---- epilogue: table twiddle (exact mod-N), scale, store -----------------
  float* yout = out + (size_t)sig*out_stride;
  #pragma unroll
  for (int r = 0; r < 8; ++r) {
    int m  = r + 8*(lane >> 4);
    int nn = lane & 15;
    int R = rowB + rt*16 + m;
    int C = colB + ct*16 + nn;
    float re = accP[r] - accQ[r];
    float im = accS[r];
    if (TWIDDLE) {
      int t = (R*C) & tw_mask;                 // ntot is a power of two
      float cs = twTab[2*t], sn = twTab[2*t + 1];
      float r2 = re*cs - im*sn; im = re*sn + im*cs; re = r2;
    }
    re *= out_scale; im *= out_scale;
    int oi = R + M*C;
    if (OUT_REAL) yout[oi] = re;
    else { yout[2*oi] = re; yout[2*oi + 1] = im; }
  }
}

// ------- host orchestration --------------------------------------------------
extern "C" void kernel_launch(void* const* d_in, const int* in_sizes, int n_in,
                              void* d_out, int out_size, void* d_ws, size_t ws_size,
                              hipStream_t stream) {
  const float* ep    = (const float*)d_in[0];
  const float* mxraw = (const float*)d_in[1];
  const float* rlog  = (const float*)d_in[2];
  const float* noise = (const float*)d_in[3];
  const float* irs   = (const float*)d_in[4];
  float* outp = (float*)d_out;
  char* base = (char*)d_ws;

  size_t off = 0;
  auto carve = [&](size_t bytes) -> size_t {
    size_t a = (off + 255) & ~(size_t)255; off = a + bytes; return a;
  };
  size_t o_t256re = carve(256*256*2), o_t256im = carve(256*256*2);
  size_t o_t128re = carve(128*128*2), o_t128im = carve(128*128*2);
  size_t o_tw32   = carve((size_t)NSAMP*2*4);
  size_t o_tw64   = carve((size_t)NPAD*2*4);
  size_t o_P      = carve((size_t)B_*NEV*sizeof(EvParams));
  size_t o_resfr  = carve((size_t)B_*NEV*NH*NFRAMES*4);
  size_t o_mx     = carve(B_*4);
  size_t o_rooms  = carve(B_*NROOMS*4);
  size_t o_res    = carve((size_t)B_*NEV*NSAMP*4);
  size_t o_ns1    = carve((size_t)NSAMP*2*4);
  size_t o_nspec  = carve((size_t)NSAMP*2*4);
  size_t o_imp    = carve((size_t)B_*NSAMP*4);
  size_t o_imps1  = carve((size_t)B_*NPAD*2*4);
  size_t o_impsp  = carve((size_t)B_*NPAD*2*4);
  size_t o_spec32 = carve((size_t)NEV*NSAMP*2*4);
  size_t o_s132   = carve((size_t)NEV*NSAMP*2*4);
  size_t o_bl     = carve((size_t)NEV*NSAMP*4);
  size_t o_loc    = carve((size_t)NEV*NSAMP*4);
  size_t o_lsum   = carve((size_t)NSAMP*4);
  size_t o_s164   = carve((size_t)NEV*NPAD*2*4);
  size_t o_Lhat   = carve((size_t)NEV*NPAD*2*4);
  size_t o_Rhat   = carve((size_t)NEV*NPAD*2*4);
  size_t o_acc    = carve((size_t)NPAD*2*4);
  size_t o_sA     = carve((size_t)NPAD*2*4);
  size_t o_conv   = carve((size_t)NPAD*4);
  size_t o_dry    = carve((size_t)NSAMP*4);
  size_t o_ds1    = carve((size_t)NPAD*2*4);
  size_t o_Dhat   = carve((size_t)NPAD*2*4);
  size_t o_What   = carve((size_t)NPAD*2*4);
  size_t o_ws1    = carve((size_t)NPAD*2*4);
  size_t o_wet    = carve((size_t)NPAD*4);
  (void)ws_size; (void)in_sizes; (void)n_in; (void)out_size; // ~98 MB total

  _Float16* t256re = (_Float16*)(base + o_t256re);
  _Float16* t256im = (_Float16*)(base + o_t256im);
  _Float16* t128re = (_Float16*)(base + o_t128re);
  _Float16* t128im = (_Float16*)(base + o_t128im);
  float* tw32   = (float*)(base + o_tw32);
  float* tw64   = (float*)(base + o_tw64);
  EvParams* P   = (EvParams*)(base + o_P);
  float* resfr  = (float*)(base + o_resfr);
  float* mx     = (float*)(base + o_mx);
  float* rooms  = (float*)(base + o_rooms);
  float* res    = (float*)(base + o_res);
  float* ns1    = (float*)(base + o_ns1);
  float* nspec  = (float*)(base + o_nspec);
  float* imp    = (float*)(base + o_imp);
  float* imps1  = (float*)(base + o_imps1);
  float* impsp  = (float*)(base + o_impsp);
  float* spec32 = (float*)(base + o_spec32);
  float* s132   = (float*)(base + o_s132);
  float* bl     = (float*)(base + o_bl);
  float* loc    = (float*)(base + o_loc);
  float* lsum   = (float*)(base + o_lsum);
  float* s164   = (float*)(base + o_s164);
  float* Lhat   = (float*)(base + o_Lhat);
  float* Rhat   = (float*)(base + o_Rhat);
  float* acc    = (float*)(base + o_acc);
  float* sA     = (float*)(base + o_sA);
  float* convs  = (float*)(base + o_conv);
  float* dry    = (float*)(base + o_dry);
  float* ds1    = (float*)(base + o_ds1);
  float* Dhat   = (float*)(base + o_Dhat);
  float* What   = (float*)(base + o_What);
  float* ws1    = (float*)(base + o_ws1);
  float* wet    = (float*)(base + o_wet);

  // flags -> template dispatch: p1-real / p1-conj(inverse) / p2 / p2-real
  auto DFT = [&](const float* i, long istr, float* o, long ostr, int M, int L,
                 const _Float16* tre, const _Float16* tim, int ireal, int kval,
                 int rs, int cs, int cj, const float* twt, int twm,
                 float scl, int oreal, int nsig) {
    dim3 g(M/32, L/64, nsig);
    if (ireal)
      k_dft_pass<true,false,true,false><<<g,256,0,stream>>>(i,o,istr,ostr,M,L,tre,tim,kval,rs,cs,twt,twm,scl);
    else if (cj)
      k_dft_pass<false,true,true,false><<<g,256,0,stream>>>(i,o,istr,ostr,M,L,tre,tim,kval,rs,cs,twt,twm,scl);
    else if (oreal)
      k_dft_pass<false,false,false,true><<<g,256,0,stream>>>(i,o,istr,ostr,M,L,tre,tim,kval,rs,cs,twt,twm,scl);
    else
      k_dft_pass<false,false,false,false><<<g,256,0,stream>>>(i,o,istr,ostr,M,L,tre,tim,kval,rs,cs,twt,twm,scl);
  };

  // prologue
  k_tables<<<384, 256, 0, stream>>>(t256re, t256im, t128re, t128im, tw32, tw64);
  k_params<<<8, 256, 0, stream>>>(ep, mxraw, rlog, P, resfr, mx, rooms);
  k_resonance<<<dim3(B_*NEV, NSAMP/256), 256, 0, stream>>>(P, resfr, res);
  // forward DFT(32768) of noise (N1=256, N2=128) — no padding, kvalid = L
  DFT(noise, 0, ns1, 0, 256, 128, t128re, t128im, 1, 128, 1, 256, 0, tw32, NSAMP-1, 1.0f, 0, 1);
  DFT(ns1, 0, nspec, 0, 128, 256, t256re, t256im, 0, 0, 256, 1, 0, tw32, 0, 1.0f, 0, 1);
  // room IRs: mix then forward DFT(65536) x4 — padded: kvalid = 128 of 256
  k_rooms_mix<<<(B_*NSAMP)/256, 256, 0, stream>>>(rooms, irs, imp);
  DFT(imp, NSAMP, imps1, (long)NPAD*2, 256, 256, t256re, t256im, 1, 128, 1, 256, 0, tw64, NPAD-1, 1.0f, 0, B_);
  DFT(imps1, (long)NPAD*2, impsp, (long)NPAD*2, 256, 256, t256re, t256im, 0, 0, 256, 1, 0, tw64, 0, 1.0f, 0, B_);

  for (int b = 0; b < B_; ++b) {
    EvParams* Pb = P + b*NEV;
    float* resb  = res + (size_t)b*NEV*NSAMP;
    float* impb  = impsp + (size_t)b*NPAD*2;
    // band-limited noise per event: full Hermitian spectrum, inverse DFT(32768)
    k_specwin<<<dim3(NEV, NSAMP/256), 256, 0, stream>>>(nspec, Pb, spec32);
    DFT(spec32, (long)NSAMP*2, s132, (long)NSAMP*2, 256, 128, t128re, t128im,
        0, 0, 1, 256, 1, tw32, NSAMP-1, 1.0f, 0, NEV);
    DFT(s132, (long)NSAMP*2, bl, NSAMP, 128, 256, t256re, t256im,
        0, 0, 256, 1, 0, tw32, 0, 1.0f/NSAMP, 1, NEV);
    // located = bl * event_window * amp ; time-domain event sum
    k_envelope<<<dim3(NEV, NSAMP/256), 256, 0, stream>>>(bl, Pb, loc);
    k_locsum<<<NSAMP/256, 256, 0, stream>>>(loc, lsum);
    // forward DFT(65536) of located and resonances (implicit zero-pad, half-K)
    DFT(loc, NSAMP, s164, (long)NPAD*2, 256, 256, t256re, t256im, 1, 128, 1, 256, 0, tw64, NPAD-1, 1.0f, 0, NEV);
    DFT(s164, (long)NPAD*2, Lhat, (long)NPAD*2, 256, 256, t256re, t256im, 0, 0, 256, 1, 0, tw64, 0, 1.0f, 0, NEV);
    DFT(resb, NSAMP, s164, (long)NPAD*2, 256, 256, t256re, t256im, 1, 128, 1, 256, 0, tw64, NPAD-1, 1.0f, 0, NEV);
    DFT(s164, (long)NPAD*2, Rhat, (long)NPAD*2, 256, 256, t256re, t256im, 0, 0, 256, 1, 0, tw64, 0, 1.0f, 0, NEV);
    // event-summed convolution spectrum, single inverse
    k_accum<<<NPAD/256, 256, 0, stream>>>(Lhat, Rhat, acc);
    DFT(acc, 0, sA, 0, 256, 256, t256re, t256im, 0, 0, 1, 256, 1, tw64, NPAD-1, 1.0f, 0, 1);
    DFT(sA, 0, convs, 0, 256, 256, t256re, t256im, 0, 0, 256, 1, 0, tw64, 0, 1.0f/NPAD, 1, 1);
    // dry (truncated), re-transform for the wet path
    k_dry<<<NSAMP/256, 256, 0, stream>>>(convs, lsum, dry);
    DFT(dry, 0, ds1, 0, 256, 256, t256re, t256im, 1, 128, 1, 256, 0, tw64, NPAD-1, 1.0f, 0, 1);
    DFT(ds1, 0, Dhat, 0, 256, 256, t256re, t256im, 0, 0, 256, 1, 0, tw64, 0, 1.0f, 0, 1);
    k_wetspec<<<NPAD/256, 256, 0, stream>>>(Dhat, impb, What);
    DFT(What, 0, ws1, 0, 256, 256, t256re, t256im, 0, 0, 1, 256, 1, tw64, NPAD-1, 1.0f, 0, 1);
    DFT(ws1, 0, wet, 0, 256, 256, t256re, t256im, 0, 0, 256, 1, 0, tw64, 0, 1.0f/NPAD, 1, 1);
    k_final<<<NSAMP/256, 256, 0, stream>>>(dry, wet, mx + b, outp + (size_t)b*NSAMP);
  }
}
